// Attention_22265110462961
// MI455X (gfx1250) — compile-verified, matrix operand order checked by reading, not run on hardware
//
#include <hip/hip_runtime.h>
#include <hip/hip_bf16.h>
#include <stdint.h>

// ---------------------------------------------------------------------------
// CDNA5 WMMA (wave32): 16x16x32 bf16 -> f32 accumulate.
// ---------------------------------------------------------------------------
typedef __attribute__((ext_vector_type(16))) __bf16 v16bf;
typedef __attribute__((ext_vector_type(8)))  __bf16 v8bf;
typedef __attribute__((ext_vector_type(4)))  __bf16 v4bf;
typedef __attribute__((ext_vector_type(8)))  float  v8f;

#define DIM 1024
#define Lq  1024
#define Bn  2
#define Hn  16
#define DHn 64
#define NTOK (Bn * Lq)   // 2048 token rows

static __device__ inline v8f wmma_bf16(v16bf a, v16bf b, v8f c) {
    return __builtin_amdgcn_wmma_f32_16x16x32_bf16(false, a, false, b,
                                                   (short)0, c, false, false);
}

// ---------------------------------------------------------------------------
// bf16 fragment load, 16-bit operand layout (ISA 05_wmma 7.12.2):
//   lane = 16*kh + idx ; elements e=0..7  hold k = kbase + 8*kh + e
//                        elements e=8..15 hold k = kbase + 16 + 8*kh + (e-8)
// Each half is 8 contiguous bf16 = 16 B = one b128 load.
// ---------------------------------------------------------------------------
static __device__ inline v16bf load_bf_frag(const __bf16* rowptr, int kbase,
                                            int kh) {
    const v8bf* p0 = reinterpret_cast<const v8bf*>(rowptr + kbase + 8 * kh);
    const v8bf* p1 = reinterpret_cast<const v8bf*>(rowptr + kbase + 16 + 8 * kh);
    v8bf lo = *p0, hi = *p1;
    return __builtin_shufflevector(lo, hi, 0, 1, 2, 3, 4, 5, 6, 7,
                                   8, 9, 10, 11, 12, 13, 14, 15);
}

// f32 source fragment (LDS probabilities): native v_cvt bf16 casts.
static __device__ inline v16bf load_frag_row_f32(const float* rowptr, int kbase,
                                                 int kh) {
    const float4* p0 = reinterpret_cast<const float4*>(rowptr + kbase + 8 * kh);
    const float4* p1 = reinterpret_cast<const float4*>(rowptr + kbase + 16 + 8 * kh);
    float4 a0 = p0[0], a1 = p0[1], a2 = p1[0], a3 = p1[1];
    v16bf r;
    r[0]  = (__bf16)a0.x; r[1]  = (__bf16)a0.y; r[2]  = (__bf16)a0.z; r[3]  = (__bf16)a0.w;
    r[4]  = (__bf16)a1.x; r[5]  = (__bf16)a1.y; r[6]  = (__bf16)a1.z; r[7]  = (__bf16)a1.w;
    r[8]  = (__bf16)a2.x; r[9]  = (__bf16)a2.y; r[10] = (__bf16)a2.z; r[11] = (__bf16)a2.w;
    r[12] = (__bf16)a3.x; r[13] = (__bf16)a3.y; r[14] = (__bf16)a3.z; r[15] = (__bf16)a3.w;
    return r;
}

// ---------------------------------------------------------------------------
// Pre-pass: f32 -> bf16 conversion (once), masked rows zeroed.
// ---------------------------------------------------------------------------
__global__ __launch_bounds__(256) void cvt_masked_kernel(
    const float* __restrict__ in, const uint8_t* __restrict__ m,
    __bf16* __restrict__ out) {
    const int i = (blockIdx.x * 256 + threadIdx.x) * 4;
    const float4 x = *reinterpret_cast<const float4*>(in + i);
    const float s = m[i >> 10] ? 0.0f : 1.0f;
    v4bf o;
    o[0] = (__bf16)(x.x * s); o[1] = (__bf16)(x.y * s);
    o[2] = (__bf16)(x.z * s); o[3] = (__bf16)(x.w * s);
    *reinterpret_cast<v4bf*>(out + i) = o;
}

__global__ __launch_bounds__(256) void cvt_plain_kernel(
    const float* __restrict__ in, __bf16* __restrict__ out) {
    const int i = (blockIdx.x * 256 + threadIdx.x) * 4;
    const float4 x = *reinterpret_cast<const float4*>(in + i);
    v4bf o;
    o[0] = (__bf16)x.x; o[1] = (__bf16)x.y;
    o[2] = (__bf16)x.z; o[3] = (__bf16)x.w;
    *reinterpret_cast<v4bf*>(out + i) = o;
}

// ---------------------------------------------------------------------------
// Software-pipelined 16x64 GEMM strip: 1 A-frag x 4 B-frags per k-step,
// fragments for step k+1 issued before consuming step k (hides L2 latency
// behind WMMA instead of s_wait_loadcnt 0).
// ---------------------------------------------------------------------------
static __device__ inline void gemm_strip_16x64(
    const __bf16* Arow, const __bf16* B0, const __bf16* B1,
    const __bf16* B2, const __bf16* B3, int kh, v8f acc[4]) {
    v16bf a  = load_bf_frag(Arow, 0, kh);
    v16bf b0 = load_bf_frag(B0, 0, kh);
    v16bf b1 = load_bf_frag(B1, 0, kh);
    v16bf b2 = load_bf_frag(B2, 0, kh);
    v16bf b3 = load_bf_frag(B3, 0, kh);
    for (int k0 = 32; k0 < DIM; k0 += 32) {
        v16bf an  = load_bf_frag(Arow, k0, kh);   // prefetch next step
        v16bf b0n = load_bf_frag(B0, k0, kh);
        v16bf b1n = load_bf_frag(B1, k0, kh);
        v16bf b2n = load_bf_frag(B2, k0, kh);
        v16bf b3n = load_bf_frag(B3, k0, kh);
        acc[0] = wmma_bf16(a, b0, acc[0]);
        acc[1] = wmma_bf16(a, b1, acc[1]);
        acc[2] = wmma_bf16(a, b2, acc[2]);
        acc[3] = wmma_bf16(a, b3, acc[3]);
        a = an; b0 = b0n; b1 = b1n; b2 = b2n; b3 = b3n;
    }
    acc[0] = wmma_bf16(a, b0, acc[0]);
    acc[1] = wmma_bf16(a, b1, acc[1]);
    acc[2] = wmma_bf16(a, b2, acc[2]);
    acc[3] = wmma_bf16(a, b3, acc[3]);
}

// ---------------------------------------------------------------------------
// Kernel 1: fused QKV projections, all-bf16 operands.
// Waves: 3 * 128 * 16 = 6144 -> 768 blocks of 8 waves.
// mat0: Qp f32 + Qpb bf16; mat1: Kpb bf16; mat2: Vt bf16 (TRANSPOSED [e][tok])
// ---------------------------------------------------------------------------
__global__ __launch_bounds__(256) void proj_qkv_kernel(
    const __bf16* __restrict__ Qb, const __bf16* __restrict__ Kb,
    const __bf16* __restrict__ Wqb, const __bf16* __restrict__ Wkb,
    const __bf16* __restrict__ Wvb,
    float* __restrict__ Qp, __bf16* __restrict__ Qpb,
    __bf16* __restrict__ Kpb, __bf16* __restrict__ Vt) {
    const int lane = threadIdx.x & 31;
    const int wave = threadIdx.x >> 5;
    const int ln = lane & 15, kh = lane >> 4;

    int gw  = blockIdx.x * 8 + wave;
    int mat = gw / 2048;                      // 0=Q,1=K,2=V
    int t   = gw - mat * 2048;
    int m0  = (t >> 4) * 16;
    int n0  = (t & 15) * 64;

    const __bf16* Ab = (mat == 0) ? Qb : Kb;
    const __bf16* Wb = (mat == 0) ? Wqb : (mat == 1) ? Wkb : Wvb;

    v8f acc[4] = {};
    gemm_strip_16x64(Ab + (size_t)(m0 + ln) * DIM,
                     Wb + (size_t)(n0 + 0  + ln) * DIM,
                     Wb + (size_t)(n0 + 16 + ln) * DIM,
                     Wb + (size_t)(n0 + 32 + ln) * DIM,
                     Wb + (size_t)(n0 + 48 + ln) * DIM, kh, acc);

    const int rb = m0 + 8 * kh;               // C/D layout: lanes 16-31 -> M=8+v
#pragma unroll
    for (int j = 0; j < 4; ++j) {
        const int col = n0 + 16 * j + ln;
        if (mat == 0) {
#pragma unroll
            for (int v = 0; v < 8; ++v) {
                const size_t idx = (size_t)(rb + v) * DIM + col;
                Qp[idx]  = acc[j][v];
                Qpb[idx] = (__bf16)acc[j][v];
            }
        } else if (mat == 1) {
#pragma unroll
            for (int v = 0; v < 8; ++v)
                Kpb[(size_t)(rb + v) * DIM + col] = (__bf16)acc[j][v];
        } else {
            v8bf p;   // transposed: fixed col, 8 consecutive tokens -> one b128
#pragma unroll
            for (int v = 0; v < 8; ++v) p[v] = (__bf16)acc[j][v];
            *reinterpret_cast<v8bf*>(Vt + (size_t)col * NTOK + rb) = p;
        }
    }
}

// ---------------------------------------------------------------------------
// Kernel 2: attention, one block per (b, h, 16-query tile).
//  A: 8 waves fill 16x1024 score strip in LDS (pipelined K-frag loads)
//  B: masked softmax, 16 lanes/row, shuffle reductions
//  C: ALL 8 waves: wave pair (w, w+4) splits the 32 k-steps of P@V for DH
//     slice w&3; upper half deposits partials in LDS, lower half reduces,
//     fuses the Qp residual and writes O f32 + Ob bf16.
// ---------------------------------------------------------------------------
__global__ __launch_bounds__(256) void attn_kernel(
    const float* __restrict__ Qp, const __bf16* __restrict__ Qpb,
    const __bf16* __restrict__ Kpb, const __bf16* __restrict__ Vt,
    const uint8_t* __restrict__ mQ, const uint8_t* __restrict__ mK,
    float* __restrict__ O, __bf16* __restrict__ Ob) {
    __shared__ float sc[16 * Lq];        // 64 KB score / probability strip
    __shared__ float part[4][32][8];     // 4 KB P@V partial accumulators

    const int qt = blockIdx.x & 63;
    const int h  = (blockIdx.x >> 6) & 15;
    const int b  = blockIdx.x >> 10;
    const int q0 = qt * 16;

    const int lane = threadIdx.x & 31;
    const int wave = threadIdx.x >> 5;
    const int ln = lane & 15, kh = lane >> 4;

    // ---- Phase A: scores = (Qh . Kh) / sqrt(DIM), pipelined ----
    const __bf16* Qrow = Qpb + ((size_t)(b * Lq + q0 + ln)) * DIM + h * DHn;
    v16bf aq0 = load_bf_frag(Qrow, 0,  kh);
    v16bf aq1 = load_bf_frag(Qrow, 32, kh);
    {
        const __bf16* Kbase = Kpb + ((size_t)(b * Lq + ln)) * DIM + h * DHn;
        const size_t krow = (size_t)16 * DIM;   // 16 key rows per tile
        v16bf bk0 = load_bf_frag(Kbase + (size_t)wave * krow, 0,  kh);
        v16bf bk1 = load_bf_frag(Kbase + (size_t)wave * krow, 32, kh);
        for (int kt = wave; kt < 64; kt += 8) {
            const int ktn = (kt + 8 < 64) ? kt + 8 : kt;   // clamped prefetch
            v16bf bk0n = load_bf_frag(Kbase + (size_t)ktn * krow, 0,  kh);
            v16bf bk1n = load_bf_frag(Kbase + (size_t)ktn * krow, 32, kh);
            v8f c = {};
            c = wmma_bf16(aq0, bk0, c);
            c = wmma_bf16(aq1, bk1, c);
            const int col = kt * 16 + ln;
#pragma unroll
            for (int v = 0; v < 8; ++v)
                sc[(8 * kh + v) * Lq + col] = c[v] * 0.03125f;  // 1/sqrt(1024)
            bk0 = bk0n; bk1 = bk1n;
        }
    }
    __syncthreads();

    // ---- Phase B: masked softmax (valid iff !mQ[q] && !mK[k]) ----
    {
        const int row = threadIdx.x >> 4;
        const int seg = threadIdx.x & 15;
        const bool qm = mQ[b * Lq + q0 + row] != 0;
        float* r = sc + row * Lq + seg * 64;
        const uint8_t* mkp = mK + b * Lq + seg * 64;

        float mloc = -3.4e38f;
        if (!qm) {
            for (int c0 = 0; c0 < 64; ++c0)
                if (!mkp[c0]) mloc = fmaxf(mloc, r[c0]);
        }
        for (int off = 1; off < 16; off <<= 1)
            mloc = fmaxf(mloc, __shfl_xor(mloc, off, 16));

        float sloc = 0.0f;
        for (int c0 = 0; c0 < 64; ++c0) {
            float p = 0.0f;
            if (!qm && !mkp[c0]) p = __expf(r[c0] - mloc);
            r[c0] = p;
            sloc += p;
        }
        for (int off = 1; off < 16; off <<= 1)
            sloc += __shfl_xor(sloc, off, 16);
        const float inv = (sloc > 0.0f) ? 1.0f / sloc : 0.0f;
        for (int c0 = 0; c0 < 64; ++c0) r[c0] *= inv;
    }
    __syncthreads();

    // ---- Phase C: O = P @ V, all 8 waves, split-k, pipelined ----
    {
        const int nt   = wave & 3;            // 16-wide DH slice
        const int half = wave >> 2;           // k-range half
        const int ktBeg = half * 16, ktEnd = ktBeg + 16;

        const float* Prow = sc + (size_t)ln * Lq;   // A lanes: M = ln
        const __bf16* Vrow =
            Vt + (size_t)(h * DHn + nt * 16 + ln) * NTOK + b * Lq;

        v8f c = {};
        v16bf a  = load_frag_row_f32(Prow, ktBeg * 32, kh);
        v16bf bv = load_bf_frag(Vrow, ktBeg * 32, kh);
        for (int kt = ktBeg; kt < ktEnd; ++kt) {
            const int ktn = (kt + 1 < ktEnd) ? kt + 1 : kt;
            v16bf an  = load_frag_row_f32(Prow, ktn * 32, kh);
            v16bf bvn = load_bf_frag(Vrow, ktn * 32, kh);
            c = wmma_bf16(a, bv, c);
            a = an; bv = bvn;
        }

        if (half == 1) {
#pragma unroll
            for (int v = 0; v < 8; ++v) part[nt][lane][v] = c[v];
        }
        __syncthreads();
        if (half == 0) {
            const int col = h * DHn + nt * 16 + ln;
#pragma unroll
            for (int v = 0; v < 8; ++v) {
                const float sum = c[v] + part[nt][lane][v];
                const size_t idx =
                    ((size_t)(b * Lq + q0 + 8 * kh + v)) * DIM + col;
                const float val = Qp[idx] + sum;   // residual: PROJECTED Q
                O[idx]  = val;
                Ob[idx] = (__bf16)val;
            }
        }
    }
}

// ---------------------------------------------------------------------------
// Kernel 3: out = O + relu(mask(O @ Wo^T)).  Pipelined 16x64 strip per wave.
// Waves: 128 * 16 = 2048 -> 256 blocks.
// ---------------------------------------------------------------------------
__global__ __launch_bounds__(256) void out_proj_kernel(
    const float* __restrict__ Oin, const __bf16* __restrict__ Obf,
    const __bf16* __restrict__ Wob, const uint8_t* __restrict__ mQ,
    float* __restrict__ out) {
    const int lane = threadIdx.x & 31;
    const int wave = threadIdx.x >> 5;
    const int ln = lane & 15, kh = lane >> 4;

    int gw = blockIdx.x * 8 + wave;
    int m0 = (gw >> 4) * 16;
    int n0 = (gw & 15) * 64;

    v8f acc[4] = {};
    gemm_strip_16x64(Obf + (size_t)(m0 + ln) * DIM,
                     Wob + (size_t)(n0 + 0  + ln) * DIM,
                     Wob + (size_t)(n0 + 16 + ln) * DIM,
                     Wob + (size_t)(n0 + 32 + ln) * DIM,
                     Wob + (size_t)(n0 + 48 + ln) * DIM, kh, acc);

    const int rb = m0 + 8 * kh;
#pragma unroll
    for (int j = 0; j < 4; ++j) {
        const int col = n0 + 16 * j + ln;
#pragma unroll
        for (int v = 0; v < 8; ++v) {
            const int row = rb + v;
            const float ff = mQ[row] ? 0.0f : fmaxf(acc[j][v], 0.0f);
            out[(size_t)row * DIM + col] = Oin[(size_t)row * DIM + col] + ff;
        }
    }
}

// ---------------------------------------------------------------------------
// Launch.  Workspace layout (48 MB total):
//   0: Qb 4MB | 4: Kb 4MB | 8: Wqb 2MB | 10: Wkb 2MB | 12: Wvb 2MB
//   14: Wob 2MB | 16: Qp f32 8MB | 24: Qpb 4MB | 28: Kpb 4MB
//   32: Vt 4MB (transposed) | 36: O f32 8MB | 44: Ob 4MB
// ---------------------------------------------------------------------------
extern "C" void kernel_launch(void* const* d_in, const int* in_sizes, int n_in,
                              void* d_out, int out_size, void* d_ws,
                              size_t ws_size, hipStream_t stream) {
    const float*   Q   = (const float*)d_in[0];
    const float*   Kin = (const float*)d_in[1];
    const uint8_t* mQ  = (const uint8_t*)d_in[2];
    const uint8_t* mK  = (const uint8_t*)d_in[3];
    const float*   Wq  = (const float*)d_in[4];
    const float*   Wk  = (const float*)d_in[5];
    const float*   Wv  = (const float*)d_in[6];
    const float*   Wo  = (const float*)d_in[7];

    char* w = (char*)d_ws;
    const size_t MB = 1024 * 1024;
    __bf16* Qb  = (__bf16*)(w + 0 * MB);
    __bf16* Kb  = (__bf16*)(w + 4 * MB);
    __bf16* Wqb = (__bf16*)(w + 8 * MB);
    __bf16* Wkb = (__bf16*)(w + 10 * MB);
    __bf16* Wvb = (__bf16*)(w + 12 * MB);
    __bf16* Wob = (__bf16*)(w + 14 * MB);
    float*  Qp  = (float*) (w + 16 * MB);
    __bf16* Qpb = (__bf16*)(w + 24 * MB);
    __bf16* Kpb = (__bf16*)(w + 28 * MB);
    __bf16* Vt  = (__bf16*)(w + 32 * MB);
    float*  O   = (float*) (w + 36 * MB);
    __bf16* Ob  = (__bf16*)(w + 44 * MB);

    cvt_masked_kernel<<<2048, 256, 0, stream>>>(Q,   mQ, Qb);
    cvt_masked_kernel<<<2048, 256, 0, stream>>>(Kin, mK, Kb);
    cvt_plain_kernel <<<1024, 256, 0, stream>>>(Wq, Wqb);
    cvt_plain_kernel <<<1024, 256, 0, stream>>>(Wk, Wkb);
    cvt_plain_kernel <<<1024, 256, 0, stream>>>(Wv, Wvb);
    cvt_plain_kernel <<<1024, 256, 0, stream>>>(Wo, Wob);

    proj_qkv_kernel<<<768, 256, 0, stream>>>(Qb, Kb, Wqb, Wkb, Wvb,
                                             Qp, Qpb, Kpb, Vt);
    attn_kernel<<<Bn * Hn * (Lq / 16), 256, 0, stream>>>(Qp, Qpb, Kpb, Vt,
                                                         mQ, mK, O, Ob);
    out_proj_kernel<<<256, 256, 0, stream>>>(O, Ob, Wob, mQ, (float*)d_out);
}
